// MaskFormerHead_84155589198520
// MI455X (gfx1250) — compile-verified, matrix operand order checked by reading, not run on hardware
//
#include <hip/hip_runtime.h>
#include <math.h>

// ---------------------------------------------------------------------------
// MaskFormer head, fused for gfx1250 (MI455X, wave32, WMMA 16x16x32 bf16).
// 16 points (64 query rows) per workgroup. Register-bounded GEMM:
// column-tile-pair outer, K inner; A reused across 2 col-tiles, B across RT
// row-tiles => 2*RT WMMAs per iteration with ~120 live VGPRs (no spills).
// ---------------------------------------------------------------------------

typedef __bf16 bf16;
typedef bf16  v16bf __attribute__((ext_vector_type(16)));
typedef float v8f   __attribute__((ext_vector_type(8)));
typedef float v4f   __attribute__((ext_vector_type(4)));

#define NTHR  256
#define NWAVE 8

// ---- WMMA fragment load (ISA 7.12.2, 16-bit A/B layout) -------------------
// lane<16 : row/col = lane,     K chunks {k0+0..7,  k0+16..23}
// lane>=16: row/col = lane-16,  K chunks {k0+8..15, k0+24..31}
__device__ __forceinline__ v16bf frag16(const bf16* m, int ld, int r0, int k0) {
  const int lane = threadIdx.x & 31;
  const bf16* p = m + (size_t)(r0 + (lane & 15)) * (size_t)ld
                    + (size_t)(k0 + ((lane & 16) ? 8 : 0));
  v16bf f;
  reinterpret_cast<v4f*>(&f)[0] = *reinterpret_cast<const v4f*>(p);
  reinterpret_cast<v4f*>(&f)[1] = *reinterpret_cast<const v4f*>(p + 16);
  return f;
}

// D = A[(RT*16) x K] * W[outF x K]^T + bias.  A in LDS (bf16), W global (bf16).
// Requires outF % 256 == 0... actually outF/16 % 16 == 0 (CT multiple of 16).
template <int RT>
__device__ __forceinline__ void gemmT(const bf16* A, int K,
                                      const bf16* W, int outF, const float* bias,
                                      float* o32, bf16* o16, int ldo, bool relu) {
  const int wave = threadIdx.x >> 5;
  const int lane = threadIdx.x & 31;
  const int CT = outF >> 4;                 // multiple of 2*NWAVE for all calls
#pragma unroll 1
  for (int ct0 = wave; ct0 < CT; ct0 += 2 * NWAVE) {
    const int ct1 = ct0 + NWAVE;
    v8f acc0[RT], acc1[RT];
    const v8f vzero = {};
#pragma unroll
    for (int r = 0; r < RT; ++r) { acc0[r] = vzero; acc1[r] = vzero; }
#pragma unroll 1
    for (int k = 0; k < K; k += 32) {
      v16bf a[RT];
#pragma unroll
      for (int r = 0; r < RT; ++r) a[r] = frag16(A, K, r * 16, k);
      v16bf b0 = frag16(W, K, ct0 * 16, k);
      v16bf b1 = frag16(W, K, ct1 * 16, k);
#pragma unroll
      for (int r = 0; r < RT; ++r)
        acc0[r] = __builtin_amdgcn_wmma_f32_16x16x32_bf16(
                    false, a[r], false, b0, (short)0, acc0[r], false, false);
#pragma unroll
      for (int r = 0; r < RT; ++r)
        acc1[r] = __builtin_amdgcn_wmma_f32_16x16x32_bf16(
                    false, a[r], false, b1, (short)0, acc1[r], false, false);
    }
#pragma unroll
    for (int half = 0; half < 2; ++half) {
      const int col = (half ? ct1 : ct0) * 16 + (lane & 15);
      const float bv = bias ? bias[col] : 0.0f;
#pragma unroll
      for (int r = 0; r < RT; ++r) {
        const v8f& acc = half ? acc1[r] : acc0[r];
        const int r0 = r * 16 + ((lane & 16) ? 8 : 0);
#pragma unroll
        for (int i = 0; i < 8; ++i) {
          float v = acc[i] + bv;
          if (relu) v = fmaxf(v, 0.0f);
          if (o32) o32[(size_t)(r0 + i) * ldo + col] = v;
          else     o16[(size_t)(r0 + i) * ldo + col] = (bf16)v;
        }
      }
    }
  }
}

__device__ __forceinline__ float wave_sum(float v) {
#pragma unroll
  for (int m = 16; m > 0; m >>= 1) v += __shfl_xor(v, m, 32);
  return v;
}

// LayerNorm of (qf + add) over 256 cols, 64 rows, one wave per row.
// perPoint: residual indexed by point (r>>2) instead of row (CA broadcast).
__device__ __forceinline__ void ln_update(float* qf, bf16* sb, const float* add,
                                          bool perPoint,
                                          const float* g, const float* b) {
  const int wave = threadIdx.x >> 5, lane = threadIdx.x & 31;
  for (int r = wave; r < 64; r += NWAVE) {
    const float* ap = add + (size_t)(perPoint ? (r >> 2) : r) * 256;
    float x[8]; float s = 0.0f;
#pragma unroll
    for (int j = 0; j < 8; ++j) { x[j] = qf[r * 256 + lane + j * 32] + ap[lane + j * 32]; s += x[j]; }
    s = wave_sum(s);
    const float mean = s * 0.00390625f;
    float vs = 0.0f;
#pragma unroll
    for (int j = 0; j < 8; ++j) { float d = x[j] - mean; vs += d * d; }
    vs = wave_sum(vs);
    const float inv = rsqrtf(vs * 0.00390625f + 1e-5f);
#pragma unroll
    for (int j = 0; j < 8; ++j) {
      const int c = lane + j * 32;
      const float y = (x[j] - mean) * inv * g[c] + b[c];
      qf[r * 256 + c] = y;
      sb[r * 256 + c] = (bf16)y;
    }
  }
}

// ---- weight f32 -> bf16 ----------------------------------------------------
__global__ __launch_bounds__(NTHR) void k_cvt(const float* __restrict__ s,
                                              bf16* __restrict__ d, int n) {
  int i = blockIdx.x * NTHR + threadIdx.x;
  if (i < n) d[i] = (bf16)s[i];
}

// ---- input projection: proj = ReLU(LN(X@W1^T+b1))@W2^T+b2  (bf16 out) -----
__global__ __launch_bounds__(NTHR) void k_input_proj(
    const float* __restrict__ feat, const bf16* __restrict__ w1,
    const float* __restrict__ b1, const float* __restrict__ g,
    const float* __restrict__ bb, const bf16* __restrict__ w2,
    const float* __restrict__ b2, bf16* __restrict__ projb) {
  __shared__ __attribute__((aligned(16))) bf16  fb[64 * 512];   // 64 KB
  __shared__ __attribute__((aligned(16))) float yf[64 * 256];   // 64 KB
  __shared__ __attribute__((aligned(16))) bf16  sb[64 * 256];   // 32 KB
  const int tid = threadIdx.x, wave = tid >> 5, lane = tid & 31;
  const size_t base = (size_t)blockIdx.x * 64;

  for (int idx = tid * 4; idx < 64 * 512; idx += NTHR * 4) {
    v4f v = *reinterpret_cast<const v4f*>(feat + base * 512 + idx);
#pragma unroll
    for (int u = 0; u < 4; ++u) fb[idx + u] = (bf16)v[u];
  }
  __syncthreads();
  gemmT<4>(fb, 512, w1, 256, b1, yf, nullptr, 256, false);
  __syncthreads();
  for (int r = wave; r < 64; r += NWAVE) {          // LN + ReLU -> bf16
    float x[8]; float s = 0.0f;
#pragma unroll
    for (int j = 0; j < 8; ++j) { x[j] = yf[r * 256 + lane + j * 32]; s += x[j]; }
    s = wave_sum(s);
    const float mean = s * 0.00390625f;
    float vs = 0.0f;
#pragma unroll
    for (int j = 0; j < 8; ++j) { float d = x[j] - mean; vs += d * d; }
    vs = wave_sum(vs);
    const float inv = rsqrtf(vs * 0.00390625f + 1e-5f);
#pragma unroll
    for (int j = 0; j < 8; ++j) {
      const int c = lane + j * 32;
      const float y = (x[j] - mean) * inv * g[c] + bb[c];
      sb[r * 256 + c] = (bf16)fmaxf(y, 0.0f);
    }
  }
  __syncthreads();
  gemmT<4>(sb, 256, w2, 256, b2, nullptr, projb + base * 256, 256, false);
}

// ---- fused decoder: 16 points (64 rows) per workgroup ---------------------
struct DecP {
  const bf16* projb; const float* qe;
  const bf16 *wsa, *wsaow, *wca, *wcaow, *wf1, *wf2, *wme1, *wme2;
  const float *sab, *saob, *n1g, *n1b, *cab, *caob, *n2g, *n2b,
              *fb1, *fb2, *n3g, *n3b, *meb1, *meb2;
  float* out;
};

__global__ __launch_bounds__(NTHR) void k_decoder(DecP P) {
  __shared__ __attribute__((aligned(16))) float qf[64 * 256];   // 64 KB q state
  __shared__ __attribute__((aligned(16))) bf16  sb[64 * 256];   // 32 KB GEMM A
  __shared__ __attribute__((aligned(16))) bf16  obb[64 * 768];  // 96 KB multi-use
  __shared__ __attribute__((aligned(16))) float sc[512 * 4];    //  8 KB attn
  __shared__ __attribute__((aligned(16))) bf16  pb[16 * 256];   //  8 KB proj
  const int tid = threadIdx.x, wave = tid >> 5, lane = tid & 31;
  const size_t base = (size_t)blockIdx.x * 16;
  float* obf = reinterpret_cast<float*>(obb);                   // f32 [64][256] (64 KB)
  float* af  = reinterpret_cast<float*>(obb + 8192);            // f32 [16][256] @16 KB
  bf16*  hb  = obb + 32768;                                     // bf16 [64][256] @64 KB

  for (int idx = tid; idx < 16 * 256; idx += NTHR)              // stage proj (bf16)
    pb[idx] = P.projb[(base + (idx >> 8)) * 256 + (idx & 255)];
  for (int idx = tid; idx < 64 * 256; idx += NTHR) {            // q = q_embed bcast
    const int qi = (idx >> 8) & 3;
    const float v = P.qe[qi * 256 + (idx & 255)];
    qf[idx] = v; sb[idx] = (bf16)v;
  }
  __syncthreads();

  for (int L = 0; L < 2; ++L) {
    // --- self-attention: QKV = q @ sa_w^T + b  -> obb[64][768] bf16
    gemmT<4>(sb, 256, P.wsa + (size_t)L * 768 * 256, 768, P.sab + L * 768,
             nullptr, obb, 768, false);
    __syncthreads();
    // --- 4x4 per-head scores + softmax (task = (p,h,qi), 512 tasks)
    for (int t = tid; t < 512; t += NTHR) {
      const int p = t >> 5, h = (t >> 2) & 7, qi = t & 3;
      const bf16* Q = obb + (size_t)(p * 4 + qi) * 768 + h * 32;
      float s[4];
#pragma unroll
      for (int ki = 0; ki < 4; ++ki) {
        const bf16* Kp = obb + (size_t)(p * 4 + ki) * 768 + 256 + h * 32;
        float d = 0.0f;
        for (int j = 0; j < 32; ++j) d += (float)Q[j] * (float)Kp[j];
        s[ki] = d * 0.17677669529663687f;   // 1/sqrt(32)
      }
      const float mx = fmaxf(fmaxf(s[0], s[1]), fmaxf(s[2], s[3]));
      float e[4]; float sum = 0.0f;
#pragma unroll
      for (int ki = 0; ki < 4; ++ki) { e[ki] = __expf(s[ki] - mx); sum += e[ki]; }
      const float rs = 1.0f / sum;
#pragma unroll
      for (int ki = 0; ki < 4; ++ki) sc[t * 4 + ki] = e[ki] * rs;
    }
    __syncthreads();
    // --- o = attn @ V -> sb (bf16)
    for (int idx = tid; idx < 64 * 256; idx += NTHR) {
      const int r = idx >> 8, c = idx & 255, p = r >> 2, qi = r & 3, h = c >> 5;
      const float* w = sc + (p * 32 + h * 4 + qi) * 4;
      float a = 0.0f;
#pragma unroll
      for (int ki = 0; ki < 4; ++ki)
        a += w[ki] * (float)obb[(size_t)(p * 4 + ki) * 768 + 512 + c];
      sb[idx] = (bf16)a;
    }
    __syncthreads();
    // --- SA out-proj -> f32, then LN1(q + a)
    gemmT<4>(sb, 256, P.wsaow + (size_t)L * 65536, 256, P.saob + L * 256,
             obf, nullptr, 256, false);
    __syncthreads();
    ln_update(qf, sb, obf, false, P.n1g + L * 256, P.n1b + L * 256);
    __syncthreads();
    // --- cross-attention: identical keys => softmax uniform => a = OW(V(proj))
    gemmT<1>(pb, 256, P.wca + (size_t)L * 768 * 256 + (size_t)512 * 256, 256,
             P.cab + L * 768 + 512, nullptr, obb, 256, false);  // v = proj@Wv^T+bv
    __syncthreads();
    gemmT<1>(obb, 256, P.wcaow + (size_t)L * 65536, 256, P.caob + L * 256,
             af, nullptr, 256, false);                          // a = v@OW^T+ob
    __syncthreads();
    ln_update(qf, sb, af, true, P.n2g + L * 256, P.n2b + L * 256);
    __syncthreads();
    // --- FFN
    gemmT<4>(sb, 256, P.wf1 + (size_t)L * 65536, 256, P.fb1 + L * 256,
             nullptr, hb, 256, true);                           // h = ReLU(...)
    __syncthreads();
    gemmT<4>(hb, 256, P.wf2 + (size_t)L * 65536, 256, P.fb2 + L * 256,
             obf, nullptr, 256, false);
    __syncthreads();
    ln_update(qf, sb, obf, false, P.n3g + L * 256, P.n3b + L * 256);
    __syncthreads();
  }
  // --- mask embed: me = ReLU(q@W1^T+b1)@W2^T+b2
  gemmT<4>(sb, 256, P.wme1, 256, P.meb1, nullptr, hb, 256, true);
  __syncthreads();
  gemmT<4>(hb, 256, P.wme2, 256, P.meb2, obf, nullptr, 256, false);
  __syncthreads();
  // --- logits = cos(me, proj)/0.07 ; out[qi*N + n]
  for (int r = wave; r < 64; r += NWAVE) {
    const int p = r >> 2, qi = r & 3;
    const float* me = obf + (size_t)r * 256;
    float sm = 0.0f, sp = 0.0f, dp = 0.0f;
#pragma unroll
    for (int j = 0; j < 8; ++j) {
      const int c = lane + j * 32;
      const float m = me[c], pr = (float)pb[p * 256 + c];
      sm += m * m; sp += pr * pr; dp += m * pr;
    }
    sm = wave_sum(sm); sp = wave_sum(sp); dp = wave_sum(dp);
    if (lane == 0) {
      const float nm = fmaxf(sqrtf(sm), 1e-12f);
      const float np = fmaxf(sqrtf(sp), 1e-12f);
      P.out[(size_t)qi * 65536 + base + p] = dp / (nm * np) * (1.0f / 0.07f);
    }
  }
}

// ---------------------------------------------------------------------------
extern "C" void kernel_launch(void* const* d_in, const int* in_sizes, int n_in,
                              void* d_out, int out_size, void* d_ws, size_t ws_size,
                              hipStream_t stream) {
  (void)in_sizes; (void)n_in; (void)out_size; (void)ws_size;
  bf16* ws = (bf16*)d_ws;
  const float* F[31];
  for (int i = 0; i < 31; ++i) F[i] = (const float*)d_in[i];

  // workspace layout (bf16 element offsets)
  enum : size_t {
    O_IP1 = 0,            O_IP2 = 131072,  O_SA  = 196608,  O_SAOW = 589824,
    O_CA  = 720896,       O_CAOW = 1114112, O_F1 = 1245184, O_F2   = 1376256,
    O_ME1 = 1507328,      O_ME2 = 1572864, O_PROJ = 1638400
  };
  struct Job { int idx; size_t off; int n; };
  const Job jobs[] = {
    {3, O_IP1, 131072}, {7, O_IP2, 65536},  {9, O_SA, 393216}, {11, O_SAOW, 131072},
    {15, O_CA, 393216}, {17, O_CAOW, 131072}, {21, O_F1, 131072}, {23, O_F2, 131072},
    {27, O_ME1, 65536}, {29, O_ME2, 65536}
  };
  for (const Job& j : jobs)
    k_cvt<<<(j.n + NTHR - 1) / NTHR, NTHR, 0, stream>>>(
        (const float*)d_in[j.idx], ws + j.off, j.n);

  k_input_proj<<<65536 / 64, NTHR, 0, stream>>>(
      (const float*)d_in[0], ws + O_IP1, F[4], F[5], F[6],
      ws + O_IP2, F[8], ws + O_PROJ);

  DecP P;
  P.projb = ws + O_PROJ; P.qe = F[2];
  P.wsa = ws + O_SA;   P.wsaow = ws + O_SAOW; P.wca = ws + O_CA; P.wcaow = ws + O_CAOW;
  P.wf1 = ws + O_F1;   P.wf2 = ws + O_F2;     P.wme1 = ws + O_ME1; P.wme2 = ws + O_ME2;
  P.sab = F[10]; P.saob = F[12]; P.n1g = F[13]; P.n1b = F[14];
  P.cab = F[16]; P.caob = F[18]; P.n2g = F[19]; P.n2b = F[20];
  P.fb1 = F[22]; P.fb2 = F[24]; P.n3g = F[25]; P.n3b = F[26];
  P.meb1 = F[28]; P.meb2 = F[30];
  P.out = (float*)d_out;
  k_decoder<<<65536 / 16, NTHR, 0, stream>>>(P);
}
